// AdaptResnetBottleneckBlock_37855841747673
// MI455X (gfx1250) — matrix operand order repaired and checked by reading, not compiled, not run on hardware
//
#include <hip/hip_runtime.h>

#define NPTS 20000
#define KNB 32
#define INFEAT 128
#define OUTF 256
#define HF 64
#define EPSB 1e-5f

typedef __attribute__((ext_vector_type(16))) _Float16 v16h;
typedef __attribute__((ext_vector_type(8)))  float    v8f;

union HFrag { v16h v; uint4 q[2]; };

// A fragment (16x32 f16, M x K): lane l -> row = l&15; halves 0..7 -> K = kbase..kbase+7,
// halves 8..15 -> K = kbase+16.., where kbase = k0 + 8*(l>=16).  Two 16B contiguous loads.
__device__ __forceinline__ v16h load_a_frag_row(const _Float16* rowp, int k0, int lane) {
  int kb = k0 + ((lane >> 4) << 3);
  HFrag f;
  f.q[0] = *(const uint4*)(rowp + kb);
  f.q[1] = *(const uint4*)(rowp + kb + 16);
  return f.v;
}
__device__ __forceinline__ v16h load_a_frag(const _Float16* base, int ld, int k0, int lane) {
  return load_a_frag_row(base + (size_t)(lane & 15) * ld, k0, lane);
}
// B fragment (32x16 f16, K x N) from column-major storage cm[col*Kd + k]:
// lane l -> col = colbase + (l&15); halves 0..7 -> K = kb..kb+7, halves 8..15 -> K = kb+16.., kb = k0 + 8*(l>=16).
__device__ __forceinline__ v16h load_b_frag(const _Float16* cm, int Kd, int colbase, int k0, int lane) {
  const _Float16* p = cm + (size_t)(colbase + (lane & 15)) * Kd + k0 + ((lane >> 4) << 3);
  HFrag f;
  f.q[0] = *(const uint4*)(p);
  f.q[1] = *(const uint4*)(p + 16);
  return f.v;
}
__device__ __forceinline__ v8f wmma16(v16h a, v16h b, v8f c) {
  return __builtin_amdgcn_wmma_f32_16x16x32_f16(false, a, false, b, (short)0, c, false, false);
}
__device__ __forceinline__ float leakyf(float v) { return v >= 0.f ? v : 0.1f * v; }

// ---------------------------------------------------------------- prep
// zero stats, fp32->f16 features, transpose+convert weights to col-major f16, zero x1 shadow row
__global__ void prep_kernel(const float* feats, const float* wu1, const float* wc0,
                            const float* wc1, const float* wu2, const float* wsc,
                            float* stats, _Float16* feat16, _Float16* wu1cm,
                            _Float16* wc0cm, _Float16* wc1cm, _Float16* wu2cm,
                            _Float16* wsccm, _Float16* x1h) {
  const size_t R0 = 3072;                 // stats zero
  const size_t R1 = R0 + (size_t)NPTS * INFEAT;  // feat convert
  const size_t R2 = R1 + HF * INFEAT;     // wu1 [128,64] -> [64][128]
  const size_t R3 = R2 + HF * HF;         // wc0 [64,64]  -> [64][64]
  const size_t R4 = R3 + HF * 192;        // wc1 [64,192] -> [192][64]
  const size_t R5 = R4 + HF * OUTF;       // wu2 [64,256] -> [256][64]
  const size_t R6 = R5 + INFEAT * OUTF;   // wsc [128,256]-> [256][128]
  const size_t R7 = R6 + HF;              // x1 shadow row
  size_t i0 = (size_t)blockIdx.x * blockDim.x + threadIdx.x;
  size_t stride = (size_t)gridDim.x * blockDim.x;
  for (size_t i = i0; i < R7; i += stride) {
    if (i < R0) { stats[i] = 0.f; }
    else if (i < R1) { size_t j = i - R0; feat16[j] = (_Float16)feats[j]; }
    else if (i < R2) { size_t j = i - R1; size_t c = j / INFEAT, k = j % INFEAT;
                       wu1cm[c * INFEAT + k] = (_Float16)wu1[k * HF + c]; }
    else if (i < R3) { size_t j = i - R2; size_t c = j / HF, k = j % HF;
                       wc0cm[c * HF + k] = (_Float16)wc0[k * HF + c]; }
    else if (i < R4) { size_t j = i - R3; size_t c = j / HF, k = j % HF;
                       wc1cm[c * HF + k] = (_Float16)wc1[k * 192 + c]; }
    else if (i < R5) { size_t j = i - R4; size_t c = j / HF, k = j % HF;
                       wu2cm[c * HF + k] = (_Float16)wu2[k * OUTF + c]; }
    else if (i < R6) { size_t j = i - R5; size_t c = j / INFEAT, k = j % INFEAT;
                       wsccm[c * INFEAT + k] = (_Float16)wsc[k * OUTF + c]; }
    else { x1h[(size_t)NPTS * HF + (i - R6)] = (_Float16)0.f; }
  }
}

// ---------------------------------------------------------------- BN finalize (scale/shift)
__global__ void bn_finalize_kernel(const float* sum, const float* ssq, const float* g,
                                   const float* b, float* scale, float* shift,
                                   int nch, float invcnt) {
  int c = blockIdx.x * blockDim.x + threadIdx.x;
  if (c < nch) {
    float m = sum[c] * invcnt;
    float v = ssq[c] * invcnt - m * m;
    float s = g[c] * rsqrtf(v + EPSB);
    scale[c] = s;
    shift[c] = b[c] - m * s;
  }
}

// ---------------------------------------------------------------- GEMM unary1: [N,128]@[128,64]
__global__ __launch_bounds__(256) void gemm_unary1_kernel(const _Float16* feat16,
                                                          const _Float16* wu1cm, float* y1raw,
                                                          float* sumA, float* ssqA) {
  __shared__ float lstat[HF][2];
  int tid = threadIdx.x, lane = tid & 31, wid = tid >> 5;
  if (tid < 128) lstat[tid >> 1][tid & 1] = 0.f;
  __syncthreads();
  int mtile = blockIdx.x * 8 + wid;
  if (mtile < NPTS / 16) {
    const _Float16* arow = feat16 + (size_t)mtile * 16 * INFEAT;
    v16h a[4];
#pragma unroll
    for (int ks = 0; ks < 4; ks++) a[ks] = load_a_frag(arow, INFEAT, ks * 32, lane);
#pragma unroll
    for (int ct = 0; ct < 4; ct++) {
      v8f acc = {};
#pragma unroll
      for (int ks = 0; ks < 4; ks++) {
        v16h b = load_b_frag(wu1cm, INFEAT, ct * 16, ks * 32, lane);
        acc = wmma16(a[ks], b, acc);
      }
      int col = ct * 16 + (lane & 15);
      int rbase = mtile * 16 + ((lane >> 4) << 3);
      float s = 0.f, q = 0.f;
#pragma unroll
      for (int j = 0; j < 8; j++) {
        float v = acc[j];
        y1raw[(size_t)(rbase + j) * HF + col] = v;
        s += v; q += v * v;
      }
      s += __shfl_xor(s, 16, 32);
      q += __shfl_xor(q, 16, 32);
      if (lane < 16) { atomicAdd(&lstat[col][0], s); atomicAdd(&lstat[col][1], q); }
    }
  }
  __syncthreads();
  if (tid < HF) { atomicAdd(&sumA[tid], lstat[tid][0]); atomicAdd(&ssqA[tid], lstat[tid][1]); }
}

// ---------------------------------------------------------------- apply bnA + leaky -> f16 x1
__global__ void apply_bn_h64_kernel(const float* in, const float* scale, const float* shift,
                                    _Float16* out16) {
  size_t i = (size_t)blockIdx.x * blockDim.x + threadIdx.x;
  if (i < (size_t)NPTS * HF) {
    int c = (int)(i & 63);
    out16[i] = (_Float16)leakyf(in[i] * scale[c] + shift[c]);
  }
}

// ---------------------------------------------------------------- pass A: bn0 statistics
__global__ __launch_bounds__(256) void passA_kernel(const _Float16* x1h, const int* inds,
                                                    const _Float16* wc0cm,
                                                    float* sum0, float* ssq0) {
  __shared__ float lstat[HF][2];
  __shared__ int lind[8][16];
  int tid = threadIdx.x, lane = tid & 31, wid = tid >> 5;
  if (tid < 128) lstat[tid >> 1][tid & 1] = 0.f;
  int tile = blockIdx.x * 8 + wid;            // 40000 tiles, grid 5000 exact
  int rbase = tile * 16;
  if (lane < 16) lind[wid][lane] = inds[rbase + lane];
  __syncthreads();
  const _Float16* rowp = x1h + (size_t)lind[wid][lane & 15] * HF;
  v16h a0 = load_a_frag_row(rowp, 0, lane);
  v16h a1 = load_a_frag_row(rowp, 32, lane);
#pragma unroll
  for (int ct = 0; ct < 4; ct++) {
    v8f acc = {};
    v16h b0 = load_b_frag(wc0cm, HF, ct * 16, 0, lane);
    v16h b1 = load_b_frag(wc0cm, HF, ct * 16, 32, lane);
    acc = wmma16(a0, b0, acc);
    acc = wmma16(a1, b1, acc);
    int col = ct * 16 + (lane & 15);
    float s = 0.f, q = 0.f;
#pragma unroll
    for (int j = 0; j < 8; j++) { float v = acc[j]; s += v; q += v * v; }
    s += __shfl_xor(s, 16, 32);
    q += __shfl_xor(q, 16, 32);
    if (lane < 16) { atomicAdd(&lstat[col][0], s); atomicAdd(&lstat[col][1], q); }
  }
  __syncthreads();
  if (tid < HF) { atomicAdd(&sum0[tid], lstat[tid][0]); atomicAdd(&ssq0[tid], lstat[tid][1]); }
}

// ---------------------------------------------------------------- pass C (heavy): one wave per point
#define PCW 4
__global__ __launch_bounds__(128) void passC_kernel(const _Float16* x1h, const int* inds,
                                                    const float* q_pts, const float* s_pts,
                                                    const _Float16* wc0cm, const _Float16* wc1cm,
                                                    const float* scale0, const float* shift0,
                                                    float* minmax, float* sum1, float* ssq1) {
  __shared__ float lstat[HF][2];
  __shared__ int lind[PCW][16];
  __shared__ float lnbx[PCW][16][3];
  __shared__ __align__(16) _Float16 lk0[PCW][16][HF];
  __shared__ float lkern[PCW][16][192];
  int tid = threadIdx.x, lane = tid & 31, wid = tid >> 5;
  if (tid < 128) lstat[tid >> 1][tid & 1] = 0.f;
  int n = blockIdx.x * PCW + wid;             // grid 5000 -> 20000 points exact
  float qx = q_pts[n * 3], qy = q_pts[n * 3 + 1], qz = q_pts[n * 3 + 2];
  int h0 = lane, h1 = lane + 32;
  float mn0 = 1e30f, mx0 = -1e30f, mn1 = 1e30f, mx1 = -1e30f;
  float s0 = 0.f, q0 = 0.f, s1 = 0.f, q1 = 0.f;
  for (int it = 0; it < 2; it++) {
    __syncthreads();                          // protect prev iter's lnbx/lkern reads
    if (lane < 16) {
      int r = n * KNB + it * 16 + lane;
      int ind = inds[r];
      lind[wid][lane] = ind;
      lnbx[wid][lane][0] = s_pts[(size_t)ind * 3]     - qx;
      lnbx[wid][lane][1] = s_pts[(size_t)ind * 3 + 1] - qy;
      lnbx[wid][lane][2] = s_pts[(size_t)ind * 3 + 2] - qz;
    }
    __syncthreads();
    const _Float16* rowp = x1h + (size_t)lind[wid][lane & 15] * HF;
    v16h a0 = load_a_frag_row(rowp, 0, lane);
    v16h a1 = load_a_frag_row(rowp, 32, lane);
    // GEMM0 (64x64) + fused bn0 + leaky -> LDS k0 tile
#pragma unroll
    for (int ct = 0; ct < 4; ct++) {
      v8f acc = {};
      v16h b0 = load_b_frag(wc0cm, HF, ct * 16, 0, lane);
      v16h b1 = load_b_frag(wc0cm, HF, ct * 16, 32, lane);
      acc = wmma16(a0, b0, acc);
      acc = wmma16(a1, b1, acc);
      int ch = ct * 16 + (lane & 15);
      float sc = scale0[ch], sh = shift0[ch];
      int rrow = (lane >> 4) << 3;
#pragma unroll
      for (int j = 0; j < 8; j++)
        lk0[wid][rrow + j][ch] = (_Float16)leakyf(acc[j] * sc + sh);
    }
    __syncthreads();
    // re-layout k0 (C layout -> A layout) through LDS, GEMM1 (64x192)
    v16h c0 = load_a_frag_row(&lk0[wid][lane & 15][0], 0, lane);
    v16h c1 = load_a_frag_row(&lk0[wid][lane & 15][0], 32, lane);
#pragma unroll
    for (int ct = 0; ct < 12; ct++) {
      v8f acc = {};
      v16h b0 = load_b_frag(wc1cm, HF, ct * 16, 0, lane);
      v16h b1 = load_b_frag(wc1cm, HF, ct * 16, 32, lane);
      acc = wmma16(c0, b0, acc);
      acc = wmma16(c1, b1, acc);
      int col = ct * 16 + (lane & 15);
      int rrow = (lane >> 4) << 3;
#pragma unroll
      for (int j = 0; j < 8; j++) lkern[wid][rrow + j][col] = acc[j];
    }
    __syncthreads();
    // contraction: msg[m,h] = kern[m, 3h..3h+2] . nb_x[m]
    for (int m = 0; m < 16; m++) {
      float x0 = lnbx[wid][m][0], x1v = lnbx[wid][m][1], x2 = lnbx[wid][m][2];
      const float* kp = &lkern[wid][m][0];
      float m0 = kp[3 * h0] * x0 + kp[3 * h0 + 1] * x1v + kp[3 * h0 + 2] * x2;
      float m1 = kp[3 * h1] * x0 + kp[3 * h1 + 1] * x1v + kp[3 * h1 + 2] * x2;
      s0 += m0; q0 += m0 * m0; mn0 = fminf(mn0, m0); mx0 = fmaxf(mx0, m0);
      s1 += m1; q1 += m1 * m1; mn1 = fminf(mn1, m1); mx1 = fmaxf(mx1, m1);
    }
  }
  size_t base = ((size_t)n * HF);
  minmax[(base + h0) * 2] = mn0; minmax[(base + h0) * 2 + 1] = mx0;
  minmax[(base + h1) * 2] = mn1; minmax[(base + h1) * 2 + 1] = mx1;
  atomicAdd(&lstat[h0][0], s0); atomicAdd(&lstat[h0][1], q0);
  atomicAdd(&lstat[h1][0], s1); atomicAdd(&lstat[h1][1], q1);
  __syncthreads();
  if (tid < HF) { atomicAdd(&sum1[tid], lstat[tid][0]); atomicAdd(&ssq1[tid], lstat[tid][1]); }
}

// ---------------------------------------------------------------- agg: bn1+leaky via min/max trick, max over K; bnc stats
__global__ void agg_kernel(const float* minmax, const float* scale1, const float* shift1,
                           float* agg, float* sumc, float* ssqc) {
  int h = threadIdx.x;
  int nb = blockIdx.x * 128;
  float sc = scale1[h], sh = shift1[h];
  float s = 0.f, q = 0.f;
  for (int i = 0; i < 128; i++) {
    int n = nb + i;
    if (n >= NPTS) break;
    float mn = minmax[((size_t)n * HF + h) * 2];
    float mx = minmax[((size_t)n * HF + h) * 2 + 1];
    float v = leakyf((sc >= 0.f ? mx : mn) * sc + sh);
    agg[(size_t)n * HF + h] = v;
    s += v; q += v * v;
  }
  atomicAdd(&sumc[h], s); atomicAdd(&ssqc[h], q);
}

// ---------------------------------------------------------------- final GEMMs: unary2 (K=64) + shortcut (K=128)
__global__ __launch_bounds__(256) void gemm_final_kernel(const _Float16* agg16,
                                                         const _Float16* feat16,
                                                         const _Float16* wu2cm, const _Float16* wsccm,
                                                         float* xoraw, float* scraw,
                                                         float* sum2, float* ssq2,
                                                         float* sumsc, float* ssqsc) {
  __shared__ float lstat[OUTF][4];
  int tid = threadIdx.x, lane = tid & 31, wid = tid >> 5;
  for (int c = tid; c < OUTF; c += blockDim.x) {
    lstat[c][0] = 0.f; lstat[c][1] = 0.f; lstat[c][2] = 0.f; lstat[c][3] = 0.f;
  }
  __syncthreads();
  int mtile = blockIdx.x * 8 + wid;
  if (mtile < NPTS / 16) {
    const _Float16* arow = agg16 + (size_t)mtile * 16 * HF;
    const _Float16* frow = feat16 + (size_t)mtile * 16 * INFEAT;
    v16h ax[2], af[4];
#pragma unroll
    for (int ks = 0; ks < 2; ks++) ax[ks] = load_a_frag(arow, HF, ks * 32, lane);
#pragma unroll
    for (int ks = 0; ks < 4; ks++) af[ks] = load_a_frag(frow, INFEAT, ks * 32, lane);
    int rbase = mtile * 16 + ((lane >> 4) << 3);
    for (int ct = 0; ct < 16; ct++) {
      int col = ct * 16 + (lane & 15);
      // unary2
      v8f acc = {};
#pragma unroll
      for (int ks = 0; ks < 2; ks++) {
        v16h b = load_b_frag(wu2cm, HF, ct * 16, ks * 32, lane);
        acc = wmma16(ax[ks], b, acc);
      }
      float s = 0.f, q = 0.f;
#pragma unroll
      for (int j = 0; j < 8; j++) {
        float v = acc[j];
        xoraw[(size_t)(rbase + j) * OUTF + col] = v;
        s += v; q += v * v;
      }
      s += __shfl_xor(s, 16, 32); q += __shfl_xor(q, 16, 32);
      if (lane < 16) { atomicAdd(&lstat[col][0], s); atomicAdd(&lstat[col][1], q); }
      // shortcut
      v8f acs = {};
#pragma unroll
      for (int ks = 0; ks < 4; ks++) {
        v16h b = load_b_frag(wsccm, INFEAT, ct * 16, ks * 32, lane);
        acs = wmma16(af[ks], b, acs);
      }
      s = 0.f; q = 0.f;
#pragma unroll
      for (int j = 0; j < 8; j++) {
        float v = acs[j];
        scraw[(size_t)(rbase + j) * OUTF + col] = v;
        s += v; q += v * v;
      }
      s += __shfl_xor(s, 16, 32); q += __shfl_xor(q, 16, 32);
      if (lane < 16) { atomicAdd(&lstat[col][2], s); atomicAdd(&lstat[col][3], q); }
    }
  }
  __syncthreads();
  for (int c = tid; c < OUTF; c += blockDim.x) {
    atomicAdd(&sum2[c], lstat[c][0]);  atomicAdd(&ssq2[c], lstat[c][1]);
    atomicAdd(&sumsc[c], lstat[c][2]); atomicAdd(&ssqsc[c], lstat[c][3]);
  }
}

// ---------------------------------------------------------------- final fused elementwise
__global__ void final_kernel(const float* xoraw, const float* scraw,
                             const float* scale2, const float* shift2,
                             const float* scalesc, const float* shiftsc, float* out) {
  size_t i = (size_t)blockIdx.x * blockDim.x + threadIdx.x;
  if (i < (size_t)NPTS * OUTF) {
    int c = (int)(i & 255);
    float v = xoraw[i] * scale2[c] + shift2[c] + scraw[i] * scalesc[c] + shiftsc[c];
    out[i] = leakyf(v);
  }
}

// ================================================================ launch
extern "C" void kernel_launch(void* const* d_in, const int* in_sizes, int n_in,
                              void* d_out, int out_size, void* d_ws, size_t ws_size,
                              hipStream_t stream) {
  const float* features = (const float*)d_in[0];
  const float* q_pts    = (const float*)d_in[1];
  const float* s_pts    = (const float*)d_in[2];
  const int*   inds     = (const int*)d_in[3];
  const float* w_unary1 = (const float*)d_in[4];
  const float* g1 = (const float*)d_in[5],  * b1 = (const float*)d_in[6];
  const float* w_conv0  = (const float*)d_in[7];
  const float* g_bn0 = (const float*)d_in[8], * b_bn0 = (const float*)d_in[9];
  const float* w_conv1  = (const float*)d_in[10];
  const float* g_bn1 = (const float*)d_in[11], * b_bn1 = (const float*)d_in[12];
  const float* g_bnc = (const float*)d_in[13], * b_bnc = (const float*)d_in[14];
  const float* w_unary2 = (const float*)d_in[15];
  const float* g2 = (const float*)d_in[16],  * b2 = (const float*)d_in[17];
  const float* w_sc     = (const float*)d_in[18];
  const float* g_sc = (const float*)d_in[19], * b_sc = (const float*)d_in[20];

  char* ws = (char*)d_ws;
  float* stats = (float*)ws;
  float *sumA = stats + 0,    *ssqA = stats + 64,   *scaleA = stats + 128,  *shiftA = stats + 192;
  float *sum0 = stats + 256,  *ssq0 = stats + 320,  *scale0 = stats + 384,  *shift0 = stats + 448;
  float *sum1 = stats + 512,  *ssq1 = stats + 576,  *scale1 = stats + 640,  *shift1 = stats + 704;
  float *sumc = stats + 768,  *ssqc = stats + 832,  *scalec = stats + 896,  *shiftc = stats + 960;
  float *sum2 = stats + 1024, *ssq2 = stats + 1280, *scale2 = stats + 1536, *shift2 = stats + 1792;
  float *sumsc = stats + 2048,*ssqsc = stats + 2304,*scalesc = stats + 2560,*shiftsc = stats + 2816;

  size_t off = 12288;
  _Float16* feat16 = (_Float16*)(ws + off); off += (size_t)NPTS * INFEAT * 2;   // 5,120,000
  _Float16* wu1cm  = (_Float16*)(ws + off); off += (size_t)HF * INFEAT * 2;     // 16,384
  _Float16* wc0cm  = (_Float16*)(ws + off); off += (size_t)HF * HF * 2;         // 8,192
  _Float16* wc1cm  = (_Float16*)(ws + off); off += (size_t)192 * HF * 2;        // 24,576
  _Float16* wu2cm  = (_Float16*)(ws + off); off += (size_t)OUTF * HF * 2;       // 32,768
  _Float16* wsccm  = (_Float16*)(ws + off); off += (size_t)OUTF * INFEAT * 2;   // 65,536
  _Float16* x1h    = (_Float16*)(ws + off); off += ((size_t)(NPTS + 1) * HF * 2 + 255) & ~(size_t)255;
  _Float16* agg16  = (_Float16*)(ws + off); off += (size_t)NPTS * HF * 2;       // 2,560,000
  float* y1raw  = (float*)(ws + off); off += (size_t)NPTS * HF * 4;             // 5,120,000
  float* minmax = (float*)(ws + off); off += (size_t)NPTS * HF * 2 * 4;         // 10,240,000
  float* aggbuf = (float*)(ws + off); off += (size_t)NPTS * HF * 4;             // 5,120,000
  float* xoraw  = (float*)(ws + off); off += (size_t)NPTS * OUTF * 4;           // 20,480,000
  float* scraw  = (float*)(ws + off); off += (size_t)NPTS * OUTF * 4;           // 20,480,000

  prep_kernel<<<4096, 256, 0, stream>>>(features, w_unary1, w_conv0, w_conv1, w_unary2, w_sc,
                                        stats, feat16, wu1cm, wc0cm, wc1cm, wu2cm, wsccm, x1h);
  gemm_unary1_kernel<<<157, 256, 0, stream>>>(feat16, wu1cm, y1raw, sumA, ssqA);
  bn_finalize_kernel<<<1, 64, 0, stream>>>(sumA, ssqA, g1, b1, scaleA, shiftA, 64, 1.f / NPTS);
  apply_bn_h64_kernel<<<5000, 256, 0, stream>>>(y1raw, scaleA, shiftA, x1h);
  passA_kernel<<<5000, 256, 0, stream>>>(x1h, inds, wc0cm, sum0, ssq0);
  bn_finalize_kernel<<<1, 64, 0, stream>>>(sum0, ssq0, g_bn0, b_bn0, scale0, shift0, 64,
                                           1.f / ((float)NPTS * KNB));
  passC_kernel<<<5000, 128, 0, stream>>>(x1h, inds, q_pts, s_pts, wc0cm, wc1cm,
                                         scale0, shift0, minmax, sum1, ssq1);
  bn_finalize_kernel<<<1, 64, 0, stream>>>(sum1, ssq1, g_bn1, b_bn1, scale1, shift1, 64,
                                           1.f / ((float)NPTS * KNB));
  agg_kernel<<<157, 64, 0, stream>>>(minmax, scale1, shift1, aggbuf, sumc, ssqc);
  bn_finalize_kernel<<<1, 64, 0, stream>>>(sumc, ssqc, g_bnc, b_bnc, scalec, shiftc, 64, 1.f / NPTS);
  apply_bn_h64_kernel<<<5000, 256, 0, stream>>>(aggbuf, scalec, shiftc, agg16);
  gemm_final_kernel<<<157, 256, 0, stream>>>(agg16, feat16, wu2cm, wsccm, xoraw, scraw,
                                             sum2, ssq2, sumsc, ssqsc);
  bn_finalize_kernel<<<1, 256, 0, stream>>>(sum2, ssq2, g2, b2, scale2, shift2, 256, 1.f / NPTS);
  bn_finalize_kernel<<<1, 256, 0, stream>>>(sumsc, ssqsc, g_sc, b_sc, scalesc, shiftsc, 256, 1.f / NPTS);
  final_kernel<<<20000, 256, 0, stream>>>(xoraw, scraw, scale2, shift2, scalesc, shiftsc,
                                          (float*)d_out);
}
